// MultiQueryCausalSelfAttention_20203526160762
// MI455X (gfx1250) — matrix operand order repaired and checked by reading, not compiled, not run on hardware
//
#include <hip/hip_runtime.h>
#include <hip/hip_bf16.h>

// ---------------------------------------------------------------------------
// CDNA5 (gfx1250) MQA causal attention, bf16 WMMA (16x16x32, f32 accum),
// async global->LDS staging (ASYNCcnt) with double-buffered GEMM tiles.
// ---------------------------------------------------------------------------

typedef __attribute__((ext_vector_type(16))) __bf16 v16bf;
typedef __attribute__((ext_vector_type(8)))  float  v8f;
typedef __attribute__((ext_vector_type(4)))  float  f32x4;
typedef int v4i_ __attribute__((vector_size(16)));   // matches builtin's V4i

union Frag16 {          // 16 bf16 = 32B = 8 VGPRs (one WMMA A/B operand)
    v16bf v;
    f32x4 f4[2];
};
union B8 {              // 8 bf16 = 16B
    f32x4  f4;
    __bf16 b[8];
};

constexpr int kB  = 4;
constexpr int kT  = 2048;
constexpr int kC  = 2048;
constexpr int kH  = 16;
constexpr int kHD = 128;
constexpr int kM  = kB * kT;     // 8192 rows

// ---- gfx1250 async global->LDS path (ASYNCcnt) ----------------------------
#if __has_builtin(__builtin_amdgcn_global_load_async_to_lds_b128) && \
    __has_builtin(__builtin_amdgcn_s_wait_asynccnt)
#define USE_ASYNC_LDS 1
#else
#define USE_ASYNC_LDS 0
#warning "gfx1250 global_load_async_to_lds/s_wait_asynccnt builtins unavailable; sync staging fallback"
#endif

__device__ __forceinline__ void copy16_g2l(const __bf16* g, __bf16* l) {
#if USE_ASYNC_LDS
    __builtin_amdgcn_global_load_async_to_lds_b128(
        (__attribute__((address_space(1))) v4i_*)g,
        (__attribute__((address_space(3))) v4i_*)l, 0, 0);
#else
    *(f32x4*)l = *(const f32x4*)g;
#endif
}
__device__ __forceinline__ void async_fence() {
#if USE_ASYNC_LDS
    __builtin_amdgcn_s_wait_asynccnt(0);
#endif
}

// ---------------------------------------------------------------------------
// GEMM: Y[M,N] = X[M,K] @ W[K,N] + bias[N]   (bf16 in, f32 acc, bf16 out)
// Block = 256 threads = 8 waves. BM=128, BN=128, BK=64, double-buffered LDS.
// Wave (wm,wn) owns a 32x64 block = 2x4 WMMA tiles -> 16 WMMA per K-step.
// ---------------------------------------------------------------------------
__global__ __launch_bounds__(256) void gemm_bias_bf16(
    const __bf16* __restrict__ X, const __bf16* __restrict__ W,
    const __bf16* __restrict__ bias, __bf16* __restrict__ Y,
    int M, int N, int K) {
    // stride 72 bf16 = 144B = 16B-aligned rows; 36 dwords -> conflict-free
    __shared__ __bf16 a_lds[2][128][72];    // rows m, cols k (row-major)
    __shared__ __bf16 wT_lds[2][128][72];   // rows n, cols k (transposed W)

    const int tid  = threadIdx.x;
    const int lane = tid & 31;
    const int wid  = tid >> 5;
    const int hl   = lane >> 4;     // half-wave select
    const int ln   = lane & 15;
    const int wm   = wid & 3;       // wave row strip (x32)
    const int wn   = wid >> 2;      // wave col strip (x64)
    const int m0   = blockIdx.y * 128;
    const int n0   = blockIdx.x * 128;

    v8f acc[2][4] = {};
    f32x4 wreg[4];

    // A tile: 128x64 bf16 = 1024 16B chunks, 4/thread, async direct to LDS
    auto stage_a = [&](int buf, int ko) {
        #pragma unroll
        for (int i = 0; i < 4; ++i) {
            const int ch = tid + i * 256;
            const int r = ch >> 3, c8 = (ch & 7) * 8;
            copy16_g2l(X + (size_t)(m0 + r) * K + ko + c8, &a_lds[buf][r][c8]);
        }
    };
    // W tile: 64x128 bf16, coalesced b128 loads into regs (no wait yet)
    auto load_w = [&](int ko) {
        #pragma unroll
        for (int i = 0; i < 4; ++i) {
            const int ch = tid + i * 256;
            const int kr = ch >> 4, c8 = (ch & 15) * 8;
            wreg[i] = *(const f32x4*)(W + (size_t)(ko + kr) * N + n0 + c8);
        }
    };
    // scatter-transpose regs -> wT (b16 stores); waits land here, not earlier
    auto scatter_w = [&](int buf) {
        #pragma unroll
        for (int i = 0; i < 4; ++i) {
            const int ch = tid + i * 256;
            const int kr = ch >> 4, c8 = (ch & 15) * 8;
            B8 d; d.f4 = wreg[i];
            #pragma unroll
            for (int e = 0; e < 8; ++e) wT_lds[buf][c8 + e][kr] = d.b[e];
        }
    };
    auto compute = [&](int buf) {
        #pragma unroll
        for (int kk = 0; kk < 2; ++kk) {
            Frag16 af[2], bfr[4];
            // A layout: lane(m,hl): e<8 -> K=hl*8+e ; e>=8 -> K=16+hl*8+(e-8)
            #pragma unroll
            for (int rt = 0; rt < 2; ++rt) {
                const __bf16* p = &a_lds[buf][wm * 32 + rt * 16 + ln][kk * 32];
                af[rt].f4[0] = *(const f32x4*)(p + hl * 8);
                af[rt].f4[1] = *(const f32x4*)(p + 16 + hl * 8);
            }
            // B layout: lane(n,hl): e<8 -> K=hl*8+e ; e>=8 -> K=16+hl*8+(e-8)
            #pragma unroll
            for (int ct = 0; ct < 4; ++ct) {
                const __bf16* p =
                    &wT_lds[buf][wn * 64 + ct * 16 + ln][kk * 32];
                bfr[ct].f4[0] = *(const f32x4*)(p + hl * 8);
                bfr[ct].f4[1] = *(const f32x4*)(p + 16 + hl * 8);
            }
            #pragma unroll
            for (int rt = 0; rt < 2; ++rt)
                #pragma unroll
                for (int ct = 0; ct < 4; ++ct)
                    acc[rt][ct] = __builtin_amdgcn_wmma_f32_16x16x32_bf16(
                        false, af[rt].v, false, bfr[ct].v,
                        (short)0, acc[rt][ct], false, false);
        }
    };

    // prologue: stage tile 0
    load_w(0);
    stage_a(0, 0);
    scatter_w(0);
    async_fence();
    __syncthreads();

    int buf = 0;
    for (int ko = 0; ko < K; ko += 64) {
        const bool has_next = (ko + 64) < K;
        if (has_next) {
            load_w(ko + 64);           // global loads in flight over compute
            stage_a(buf ^ 1, ko + 64); // async copies in flight over compute
        }
        compute(buf);
        if (has_next) {
            scatter_w(buf ^ 1);
            async_fence();
        }
        __syncthreads();
        buf ^= 1;
    }

    // epilogue: C/D layout row = r + 8*hl, col = ln
    #pragma unroll
    for (int rt = 0; rt < 2; ++rt)
        #pragma unroll
        for (int ct = 0; ct < 4; ++ct) {
            const int col = n0 + wn * 64 + ct * 16 + ln;
            const float bv = (float)bias[col];
            #pragma unroll
            for (int r = 0; r < 8; ++r) {
                const int row = m0 + wm * 32 + rt * 16 + r + 8 * hl;
                Y[(size_t)row * N + col] = (__bf16)(acc[rt][ct][r] + bv);
            }
        }
}

// ---------------------------------------------------------------------------
// Flash MQA: one workgroup per (b, h, 128 q-rows); wave w owns 16 q rows.
// Key/value streamed in 32-row tiles; online softmax; O in f32 registers.
// ---------------------------------------------------------------------------
__global__ __launch_bounds__(256) void flash_mqa(
    const __bf16* __restrict__ Q, const __bf16* __restrict__ KV,
    __bf16* __restrict__ Yo) {
    __shared__ __bf16 k_lds[32][136];      // K tile row-major (s, d); 272B stride
    __shared__ __bf16 vT_lds[128][40];     // V tile transposed (d, s); 80B stride
    __shared__ __bf16 p_lds[8][16][40];    // per-wave P tile (m, s)

    const int tid  = threadIdx.x;
    const int lane = tid & 31;
    const int wid  = tid >> 5;
    const int hl   = lane >> 4;
    const int ln   = lane & 15;

    const int wg = blockIdx.x;
    const int qt = wg & 15;                // T/128 = 16 q-tiles
    const int h  = (wg >> 4) & 15;         // H = 16
    const int b  = wg >> 8;
    const int t0 = qt * 128 + wid * 16;    // this wave's first q row

    // Q fragments (16 rows x 128 d), resident all kernel. A layout.
    Frag16 qf[4];
    {
        const __bf16* qrow = Q + (size_t)(b * kT + t0 + ln) * kC + h * kHD;
        #pragma unroll
        for (int kk = 0; kk < 4; ++kk) {
            qf[kk].f4[0] = *(const f32x4*)(qrow + kk * 32 + hl * 8);
            qf[kk].f4[1] = *(const f32x4*)(qrow + kk * 32 + 16 + hl * 8);
        }
    }

    v8f o_acc[8] = {};
    float m_i[8], l_i[8];
    #pragma unroll
    for (int r = 0; r < 8; ++r) { m_i[r] = -3.0e38f; l_i[r] = 0.0f; }

    const float scale  = 0.08838834764831845f;   // 1/sqrt(128)
    const int   ntiles = (qt + 1) * 4;           // causal key coverage

    for (int kt = 0; kt < ntiles; ++kt) {
        const int s0 = kt * 32;
        // K tile: 32x128 = 512 16B chunks, 2/thread, async to LDS
        f32x4 vreg[2];
        #pragma unroll
        for (int i = 0; i < 2; ++i) {
            const int ch = tid + i * 256;
            const int r = ch >> 4, c8 = (ch & 15) * 8;
            const __bf16* kvrow = KV + (size_t)(b * kT + s0 + r) * (2 * kHD);
            copy16_g2l(kvrow + c8, &k_lds[r][c8]);
            vreg[i] = *(const f32x4*)(kvrow + kHD + c8);   // V loads in flight
        }
        #pragma unroll
        for (int i = 0; i < 2; ++i) {
            const int ch = tid + i * 256;
            const int r = ch >> 4, c8 = (ch & 15) * 8;
            B8 d; d.f4 = vreg[i];
            #pragma unroll
            for (int e = 0; e < 8; ++e) vT_lds[c8 + e][r] = d.b[e];
        }
        async_fence();
        __syncthreads();

        // S = Q @ K^T : 2 col-tiles of 16 keys, 4 k-steps over d=128
        v8f s_acc[2] = {};
        #pragma unroll
        for (int kk = 0; kk < 4; ++kk)
            #pragma unroll
            for (int sc = 0; sc < 2; ++sc) {
                Frag16 bk;
                const __bf16* kp = &k_lds[sc * 16 + ln][kk * 32];
                bk.f4[0] = *(const f32x4*)(kp + hl * 8);
                bk.f4[1] = *(const f32x4*)(kp + 16 + hl * 8);
                s_acc[sc] = __builtin_amdgcn_wmma_f32_16x16x32_bf16(
                    false, qf[kk].v, false, bk.v, (short)0, s_acc[sc],
                    false, false);
            }

        // online softmax; C/D layout: row = r + 8*hl, col = s0 + sc*16 + ln
        #pragma unroll
        for (int r = 0; r < 8; ++r) {
            const int trow = t0 + r + 8 * hl;
            float s0v = (s0 + ln      > trow) ? -3.0e38f : s_acc[0][r] * scale;
            float s1v = (s0 + 16 + ln > trow) ? -3.0e38f : s_acc[1][r] * scale;
            float mx = fmaxf(s0v, s1v);
            #pragma unroll
            for (int off = 1; off < 16; off <<= 1)
                mx = fmaxf(mx, __shfl_xor(mx, off, 32));
            const float mnew = fmaxf(m_i[r], mx);
            const float p0 = __expf(s0v - mnew);
            const float p1 = __expf(s1v - mnew);
            float rs = p0 + p1;
            #pragma unroll
            for (int off = 1; off < 16; off <<= 1)
                rs += __shfl_xor(rs, off, 32);
            const float corr = __expf(m_i[r] - mnew);
            l_i[r] = l_i[r] * corr + rs;
            m_i[r] = mnew;
            #pragma unroll
            for (int oc = 0; oc < 8; ++oc) o_acc[oc][r] *= corr;
            p_lds[wid][r + 8 * hl][ln]      = (__bf16)p0;
            p_lds[wid][r + 8 * hl][16 + ln] = (__bf16)p1;
        }
        __syncthreads();   // D-layout f32 -> A-layout bf16 via LDS

        // O += P @ V : one WMMA (K=32) per 16-wide output slice
        Frag16 pf;
        {
            const __bf16* pp = &p_lds[wid][ln][0];
            pf.f4[0] = *(const f32x4*)(pp + hl * 8);
            pf.f4[1] = *(const f32x4*)(pp + 16 + hl * 8);
        }
        #pragma unroll
        for (int oc = 0; oc < 8; ++oc) {
            Frag16 bv;
            const __bf16* vp = &vT_lds[oc * 16 + ln][0];
            bv.f4[0] = *(const f32x4*)(vp + hl * 8);
            bv.f4[1] = *(const f32x4*)(vp + 16 + hl * 8);
            o_acc[oc] = __builtin_amdgcn_wmma_f32_16x16x32_bf16(
                false, pf.v, false, bv.v, (short)0, o_acc[oc], false, false);
        }
        __syncthreads();   // protect k_lds / vT_lds before next tile
    }

    // y = O / l, bf16 to workspace (B,T,C) layout
    #pragma unroll
    for (int r = 0; r < 8; ++r) {
        const float inv = 1.0f / l_i[r];
        const size_t rowoff =
            (size_t)(b * kT + t0 + r + 8 * hl) * kC + h * kHD;
        #pragma unroll
        for (int oc = 0; oc < 8; ++oc)
            Yo[rowoff + oc * 16 + ln] = (__bf16)(o_acc[oc][r] * inv);
    }
}

// ---------------------------------------------------------------------------
extern "C" void kernel_launch(void* const* d_in, const int* in_sizes, int n_in,
                              void* d_out, int out_size, void* d_ws, size_t ws_size,
                              hipStream_t stream) {
    (void)in_sizes; (void)n_in; (void)out_size; (void)ws_size;
    const __bf16* x   = (const __bf16*)d_in[0];
    const __bf16* Wq  = (const __bf16*)d_in[1];
    const __bf16* bq  = (const __bf16*)d_in[2];
    const __bf16* Wkv = (const __bf16*)d_in[3];
    const __bf16* bkv = (const __bf16*)d_in[4];
    const __bf16* Wp  = (const __bf16*)d_in[5];
    const __bf16* bp  = (const __bf16*)d_in[6];
    __bf16* out = (__bf16*)d_out;

    char* ws = (char*)d_ws;
    __bf16* q_ws  = (__bf16*)ws;                                  // 8192*2048 bf16
    __bf16* kv_ws = (__bf16*)(ws + (size_t)kM * kC * 2);          // 8192*256 bf16
    __bf16* y_ws  = (__bf16*)(ws + (size_t)kM * kC * 2
                                 + (size_t)kM * 2 * kHD * 2);     // 8192*2048 bf16

    const dim3 blk(256);
    // q = x @ Wq + bq
    gemm_bias_bf16<<<dim3(kC / 128, kM / 128), blk, 0, stream>>>(
        x, Wq, bq, q_ws, kM, kC, kC);
    // kv = x @ Wkv + bkv
    gemm_bias_bf16<<<dim3((2 * kHD) / 128, kM / 128), blk, 0, stream>>>(
        x, Wkv, bkv, kv_ws, kM, 2 * kHD, kC);
    // flash attention: B * H * (T/128) workgroups
    flash_mqa<<<dim3(kB * kH * (kT / 128)), blk, 0, stream>>>(q_ws, kv_ws, y_ws);
    // out = y @ Wp + bp
    gemm_bias_bf16<<<dim3(kC / 128, kM / 128), blk, 0, stream>>>(
        y_ws, Wp, bp, out, kM, kC, kC);
}
